// HypergraphAttentionLayer_9466107920683
// MI455X (gfx1250) — matrix operand order repaired and checked by reading, not compiled, not run on hardware
//
#include <hip/hip_runtime.h>
#include <math.h>

// Problem constants (fixed by the reference)
#define Nn 8192
#define Ee 2048
#define Dd 256
#define HD 64
#define NHEADS 4

typedef __attribute__((ext_vector_type(16))) _Float16 v16h;
typedef __attribute__((ext_vector_type(8)))  _Float16 v8h;
typedef __attribute__((ext_vector_type(8)))  float    v8f;
typedef __attribute__((ext_vector_type(4)))  int      v4i;

#define HAS_ASYNC_LDS __has_builtin(__builtin_amdgcn_global_load_async_to_lds_b128)

__device__ __forceinline__ v8f wmma_f16(v16h a, v16h b, v8f c) {
  // (neg_a, A, neg_b, B, c_mod, C, reuse_a, reuse_b)
  return __builtin_amdgcn_wmma_f32_16x16x32_f16(false, a, false, b, (short)0, c,
                                                false, false);
}

__device__ __forceinline__ v16h cat8(v8h lo, v8h hi) {
  return __builtin_shufflevector(lo, hi, 0, 1, 2, 3, 4, 5, 6, 7,
                                 8, 9, 10, 11, 12, 13, 14, 15);
}

// ---------------------------------------------------------------------------
// Elementwise fp32 -> fp16
// ---------------------------------------------------------------------------
__global__ __launch_bounds__(256)
void cvt_f32_f16_kernel(const float* __restrict__ in, _Float16* __restrict__ out,
                        int n) {
  const int i = blockIdx.x * 256 + threadIdx.x;
  if (i < n) out[i] = (_Float16)in[i];
}

// ---------------------------------------------------------------------------
// C[M x Dout] = A[M x K] * W[Dout x K]^T + bias, all-f16 operands, f32 accum.
// Optional outputs: C32 (fp32), C16 (fp16 row-major), C16T (fp16 transposed,
// leading dim ldT). Block: 128 threads (4 waves), 64x64 tile.
// Weight tile (shared by all 4 waves) staged in LDS with async-to-LDS copies.
// ---------------------------------------------------------------------------
__global__ __launch_bounds__(128)
void gemm16_kernel(const _Float16* __restrict__ A, const _Float16* __restrict__ W,
                   const float* __restrict__ bias,
                   float* __restrict__ C32, _Float16* __restrict__ C16,
                   _Float16* __restrict__ C16T, int ldT, int K, int Dout) {
  __shared__ _Float16 wtile[2][64 * 32];   // 64 output-cols x 32 k, double buf
  const int wave = threadIdx.x >> 5;
  const int lane = threadIdx.x & 31;
  const int r    = lane & 15;
  const int half = lane >> 4;
  const int rowbase = blockIdx.x * 64 + wave * 16;
  const int colbase = blockIdx.y * 64;

  // stage one 64x32 f16 weight tile into LDS (4 KB): 256 16-byte chunks,
  // 128 threads -> 2 chunks per thread.
  auto stage = [&](int buf, int k0) {
#pragma unroll
    for (int i = 0; i < 2; ++i) {
      const int idx = (int)threadIdx.x + 128 * i;
      const int rr  = idx >> 2;     // 0..63 : output col within tile
      const int ch  = idx & 3;      // 16-byte chunk within the 64-byte row
      const _Float16* g = W + (size_t)(colbase + rr) * K + k0 + ch * 8;
      _Float16* l = &wtile[buf][rr * 32 + ch * 8];
#if HAS_ASYNC_LDS
      __builtin_amdgcn_global_load_async_to_lds_b128(
          (__attribute__((address_space(1))) v4i*)g,
          (__attribute__((address_space(3))) v4i*)l, 0, 0);
#else
      *(v8h*)l = *(const v8h*)g;
#endif
    }
  };

  v8f acc[4] = {v8f{}, v8f{}, v8f{}, v8f{}};
  const _Float16* arow = A + (size_t)(rowbase + r) * K;

  stage(0, 0);
  int buf = 0;
  for (int k0 = 0; k0 < K; k0 += 32) {
#if HAS_ASYNC_LDS
#if __has_builtin(__builtin_amdgcn_s_wait_asynccnt)
    __builtin_amdgcn_s_wait_asynccnt(0);
#else
    asm volatile("s_wait_asynccnt 0x0" ::: "memory");
#endif
#endif
    __syncthreads();
    if (k0 + 32 < K) stage(buf ^ 1, k0 + 32);

    // A fragment: 16x32, ISA 16-bit A layout -> two contiguous v8h loads
    const v8h alo = *(const v8h*)(arow + k0 + half * 8);
    const v8h ahi = *(const v8h*)(arow + k0 + 16 + half * 8);
    const v16h af = cat8(alo, ahi);

#pragma unroll
    for (int t = 0; t < 4; ++t) {
      // B fragment: ISA 16-bit B layout -> two contiguous v8h from LDS
      const v16h bf =
          cat8(*(const v8h*)&wtile[buf][(t * 16 + r) * 32 + half * 8],
               *(const v8h*)&wtile[buf][(t * 16 + r) * 32 + 16 + half * 8]);
      acc[t] = wmma_f16(af, bf, acc[t]);
    }
    buf ^= 1;
  }

#pragma unroll
  for (int t = 0; t < 4; ++t) {
    const int col = colbase + t * 16 + r;
    const float bv = bias[col];
    float ov[8];
#pragma unroll
    for (int v = 0; v < 8; ++v) ov[v] = acc[t][v] + bv;
    if (C32) {
#pragma unroll
      for (int v = 0; v < 8; ++v)
        C32[(size_t)(rowbase + v + 8 * half) * Dout + col] = ov[v];
    }
    if (C16) {
#pragma unroll
      for (int v = 0; v < 8; ++v)
        C16[(size_t)(rowbase + v + 8 * half) * Dout + col] = (_Float16)ov[v];
    }
    if (C16T) {
      // rows are contiguous over the accumulator index -> one v8h store
      v8h hv;
#pragma unroll
      for (int v = 0; v < 8; ++v) hv[v] = (_Float16)ov[v];
      *(v8h*)&C16T[(size_t)col * ldT + rowbase + 8 * half] = hv;
    }
  }
}

// ---------------------------------------------------------------------------
// Flash-style masked MHA. One wave per (16-query tile, head).
// Q,K: f16 [L x 256] row-major. Vt: f16 [256 x Lk] (transposed). Out: f16.
// maskMode 0: blocked where H[key, query]==0   (node->edge, mask = H.T==0)
// maskMode 1: blocked where H[query, key]==0   (edge->node, mask = H==0)
// ---------------------------------------------------------------------------
__global__ __launch_bounds__(32)
void attn_kernel(const _Float16* __restrict__ Q, const _Float16* __restrict__ Km,
                 const _Float16* __restrict__ Vt, const float* __restrict__ Hmat,
                 _Float16* __restrict__ Out, int Lk, int ldV, int maskMode) {
  __shared__ _Float16 pbuf[16 * 32];   // P tile: D-layout -> A-layout staging
  const int lane = threadIdx.x & 31;
  const int r    = lane & 15;
  const int half = lane >> 4;
  const int qb   = blockIdx.x * 16;
  const int hoff = blockIdx.y * HD;

  // Q fragments (16 queries x 64 dims = two 16x32 A fragments), resident.
  const _Float16* qrow = Q + (size_t)(qb + r) * Dd + hoff;
  const v16h qa0 = cat8(*(const v8h*)(qrow + half * 8),
                        *(const v8h*)(qrow + 16 + half * 8));
  const v16h qa1 = cat8(*(const v8h*)(qrow + 32 + half * 8),
                        *(const v8h*)(qrow + 48 + half * 8));

  float mrow[8], lsum[8];
  v8f acc[4] = {v8f{}, v8f{}, v8f{}, v8f{}};
#pragma unroll
  for (int v = 0; v < 8; ++v) { mrow[v] = -3.0e38f; lsum[v] = 0.0f; }

  for (int kb = 0; kb < Lk; kb += 32) {
    // ---- S = Q * K^T for 32 keys (two 16x16 tiles, 2 WMMA each) ----
    v8f s[2];
#pragma unroll
    for (int t = 0; t < 2; ++t) {
      const _Float16* krow = Km + (size_t)(kb + t * 16 + r) * Dd + hoff;
      const v16h b0 = cat8(*(const v8h*)(krow + half * 8),
                           *(const v8h*)(krow + 16 + half * 8));
      const v16h b1 = cat8(*(const v8h*)(krow + 32 + half * 8),
                           *(const v8h*)(krow + 48 + half * 8));
      v8f c = {};
      c = wmma_f16(qa0, b0, c);
      c = wmma_f16(qa1, b1, c);
      s[t] = c;
    }

    // ---- scale + incidence mask (NEG = -1e9, matching reference) ----
#pragma unroll
    for (int t = 0; t < 2; ++t) {
      const int kn = kb + t * 16 + r;
#pragma unroll
      for (int v = 0; v < 8; ++v) {
        const int qg = qb + v + 8 * half;
        const size_t hidx = maskMode ? ((size_t)qg * Ee + kn)
                                     : ((size_t)kn * Ee + qg);
        const float hv = Hmat[hidx];
        const float sv = s[t][v] * 0.125f;       // 1/sqrt(64)
        s[t][v] = (hv != 0.0f) ? sv : -1.0e9f;
      }
    }

    // ---- online softmax; rows live at (vgpr v, half), cols across 16 lanes
#pragma unroll
    for (int v = 0; v < 8; ++v) {
      float loc = fmaxf(s[0][v], s[1][v]);
      loc = fmaxf(loc, __shfl_xor(loc, 1, 32));
      loc = fmaxf(loc, __shfl_xor(loc, 2, 32));
      loc = fmaxf(loc, __shfl_xor(loc, 4, 32));
      loc = fmaxf(loc, __shfl_xor(loc, 8, 32));
      const float nm = fmaxf(mrow[v], loc);
      const float cf = __expf(mrow[v] - nm);
      mrow[v] = nm;
      const float e0 = __expf(s[0][v] - nm);
      const float e1 = __expf(s[1][v] - nm);
      float rs = e0 + e1;
      rs += __shfl_xor(rs, 1, 32);
      rs += __shfl_xor(rs, 2, 32);
      rs += __shfl_xor(rs, 4, 32);
      rs += __shfl_xor(rs, 8, 32);
      lsum[v] = lsum[v] * cf + rs;
#pragma unroll
      for (int t4 = 0; t4 < 4; ++t4) acc[t4][v] *= cf;
      pbuf[(v + 8 * half) * 32 + r]      = (_Float16)e0;  // P[row][key_local]
      pbuf[(v + 8 * half) * 32 + 16 + r] = (_Float16)e1;
    }
    __syncthreads();

    // ---- P as A operand (16 queries x 32 keys): two contiguous v8h ----
    const v16h pa = cat8(*(const v8h*)&pbuf[r * 32 + half * 8],
                         *(const v8h*)&pbuf[r * 32 + 16 + half * 8]);
    __syncthreads();

    // ---- acc += P * V  (4 col-tiles of the 64-wide head) ----
#pragma unroll
    for (int t4 = 0; t4 < 4; ++t4) {
      const int col = hoff + t4 * 16 + r;
      const v16h vb = cat8(*(const v8h*)&Vt[(size_t)col * ldV + kb + half * 8],
                           *(const v8h*)&Vt[(size_t)col * ldV + kb + 16 + half * 8]);
      acc[t4] = wmma_f16(pa, vb, acc[t4]);
    }
  }

  // ---- normalize + store (f16, feeds the out-projection GEMM) ----
#pragma unroll
  for (int t4 = 0; t4 < 4; ++t4) {
    const int col = hoff + t4 * 16 + r;
#pragma unroll
    for (int v = 0; v < 8; ++v) {
      const int row = qb + v + 8 * half;
      Out[(size_t)row * Dd + col] = (_Float16)(acc[t4][v] / lsum[v]);
    }
  }
}

// ---------------------------------------------------------------------------
// Out = LayerNorm(X + Y) * g + b, row-wise over D=256. One wave per row.
// Optional f16 mirror of the output (feeds later GEMMs).
// ---------------------------------------------------------------------------
__global__ __launch_bounds__(256)
void ln_residual_kernel(const float* __restrict__ X, const float* __restrict__ Y,
                        const float* __restrict__ g, const float* __restrict__ b,
                        float* __restrict__ Out, _Float16* __restrict__ Out16) {
  const int wave = threadIdx.x >> 5;
  const int lane = threadIdx.x & 31;
  const int row  = blockIdx.x * 8 + wave;
  const float* xr = X + (size_t)row * Dd;
  const float* yr = Y + (size_t)row * Dd;
  float vals[8];
  float s = 0.0f;
#pragma unroll
  for (int j = 0; j < 8; ++j) {
    const int idx = lane + 32 * j;
    vals[j] = xr[idx] + yr[idx];
    s += vals[j];
  }
#pragma unroll
  for (int m = 1; m < 32; m <<= 1) s += __shfl_xor(s, m, 32);
  const float mu = s * (1.0f / 256.0f);
  float vs = 0.0f;
#pragma unroll
  for (int j = 0; j < 8; ++j) { const float d = vals[j] - mu; vs += d * d; }
#pragma unroll
  for (int m = 1; m < 32; m <<= 1) vs += __shfl_xor(vs, m, 32);
  const float rstd = rsqrtf(vs * (1.0f / 256.0f) + 1e-5f);
  float* outr = Out + (size_t)row * Dd;
#pragma unroll
  for (int j = 0; j < 8; ++j) {
    const int idx = lane + 32 * j;
    const float o = (vals[j] - mu) * rstd * g[idx] + b[idx];
    outr[idx] = o;
    if (Out16) Out16[(size_t)row * Dd + idx] = (_Float16)o;
  }
}

// ---------------------------------------------------------------------------
extern "C" void kernel_launch(void* const* d_in, const int* in_sizes, int n_in,
                              void* d_out, int out_size, void* d_ws,
                              size_t ws_size, hipStream_t stream) {
  (void)in_sizes; (void)n_in; (void)out_size; (void)ws_size;
  const float* x0        = (const float*)d_in[0];
  const float* x1        = (const float*)d_in[1];
  const float* Hm        = (const float*)d_in[2];
  const float* node_w    = (const float*)d_in[3];
  const float* node_b    = (const float*)d_in[4];
  const float* edge_w    = (const float*)d_in[5];
  const float* edge_b    = (const float*)d_in[6];
  const float* n2e_in_w  = (const float*)d_in[7];
  const float* n2e_in_b  = (const float*)d_in[8];
  const float* n2e_out_w = (const float*)d_in[9];
  const float* n2e_out_b = (const float*)d_in[10];
  const float* e2n_in_w  = (const float*)d_in[11];
  const float* e2n_in_b  = (const float*)d_in[12];
  const float* e2n_out_w = (const float*)d_in[13];
  const float* e2n_out_b = (const float*)d_in[14];
  const float* n1g       = (const float*)d_in[15];
  const float* n1b       = (const float*)d_in[16];
  const float* n2g       = (const float*)d_in[17];
  const float* n2b       = (const float*)d_in[18];

  float* out = (float*)d_out;
  const size_t ND = (size_t)Nn * Dd;   // 2M
  const size_t ED = (size_t)Ee * Dd;   // 0.5M
  const size_t WD = (size_t)Dd * Dd;   // 64K (one DxD weight)

  // ---- workspace carve-up (~43 MB) ----
  float* x0p = (float*)d_ws;       // [ND] fp32 residual for LN2
  float* x1p = x0p + ND;           // [ED] fp32 residual for LN1
  float* tmp = x1p + ED;           // [ND] fp32 out-proj result before LN
  _Float16* h = (_Float16*)(tmp + ND);
  _Float16* x0h   = h;             h += ND;      // f16 inputs
  _Float16* x1h   = h;             h += ED;
  _Float16* w_nd  = h;             h += WD;      // f16 weights
  _Float16* w_ed  = h;             h += WD;
  _Float16* w_i1  = h;             h += 3 * WD;
  _Float16* w_o1  = h;             h += WD;
  _Float16* w_i2  = h;             h += 3 * WD;
  _Float16* w_o2  = h;             h += WD;
  _Float16* x0ph  = h;             h += ND;      // f16 activations
  _Float16* x1ph  = h;             h += ED;
  _Float16* q1h   = h;             h += ED;
  _Float16* k1h   = h;             h += ND;      // reused as q2
  _Float16* vt    = h;             h += ND;      // v1t; reused as k2 + v2t
  _Float16* atth  = h;             h += ND;      // attention outputs
  _Float16* x1uh  = h;             h += ED;      // f16 mirror of x_1_u

  float* x1u = out + ND;           // x_1_u: output #2 and MHA2's KV source

  const dim3 cblk(256);
  const dim3 gblk(128);
  auto cvt = [&](const float* src, _Float16* dst, int n) {
    cvt_f32_f16_kernel<<<dim3((n + 255) / 256), cblk, 0, stream>>>(src, dst, n);
  };

  // fp32 -> fp16 staging (inputs + weights, one pass each)
  cvt(x0, x0h, (int)ND);
  cvt(x1, x1h, (int)ED);
  cvt(node_w, w_nd, (int)WD);
  cvt(edge_w, w_ed, (int)WD);
  cvt(n2e_in_w, w_i1, (int)(3 * WD));
  cvt(n2e_out_w, w_o1, (int)WD);
  cvt(e2n_in_w, w_i2, (int)(3 * WD));
  cvt(e2n_out_w, w_o2, (int)WD);

  // Input projections (f32 for residuals + f16 for downstream GEMMs)
  gemm16_kernel<<<dim3(Nn / 64, Dd / 64), gblk, 0, stream>>>(
      x0h, w_nd, node_b, x0p, x0ph, nullptr, 0, Dd, Dd);
  gemm16_kernel<<<dim3(Ee / 64, Dd / 64), gblk, 0, stream>>>(
      x1h, w_ed, edge_b, x1p, x1ph, nullptr, 0, Dd, Dd);

  // MHA1 (node -> edge): Q from edges, K/V from nodes; V written transposed
  gemm16_kernel<<<dim3(Ee / 64, Dd / 64), gblk, 0, stream>>>(
      x1ph, w_i1, n2e_in_b, nullptr, q1h, nullptr, 0, Dd, Dd);
  gemm16_kernel<<<dim3(Nn / 64, Dd / 64), gblk, 0, stream>>>(
      x0ph, w_i1 + WD, n2e_in_b + Dd, nullptr, k1h, nullptr, 0, Dd, Dd);
  gemm16_kernel<<<dim3(Nn / 64, Dd / 64), gblk, 0, stream>>>(
      x0ph, w_i1 + 2 * WD, n2e_in_b + 2 * Dd, nullptr, nullptr, vt, Nn, Dd, Dd);
  attn_kernel<<<dim3(Ee / 16, NHEADS), dim3(32), 0, stream>>>(
      q1h, k1h, vt, Hm, atth, Nn, Nn, 0);
  gemm16_kernel<<<dim3(Ee / 64, Dd / 64), gblk, 0, stream>>>(
      atth, w_o1, n2e_out_b, tmp, nullptr, nullptr, 0, Dd, Dd);
  ln_residual_kernel<<<dim3(Ee / 8), cblk, 0, stream>>>(
      x1p, tmp, n1g, n1b, x1u, x1uh);

  // MHA2 (edge -> node): Q from nodes, K/V from updated edges (x1u)
  _Float16* q2h = k1h;             // reuse
  _Float16* k2h = vt;              // reuse (ED)
  _Float16* v2t = vt + ED;         // reuse (ED)
  gemm16_kernel<<<dim3(Nn / 64, Dd / 64), gblk, 0, stream>>>(
      x0ph, w_i2, e2n_in_b, nullptr, q2h, nullptr, 0, Dd, Dd);
  gemm16_kernel<<<dim3(Ee / 64, Dd / 64), gblk, 0, stream>>>(
      x1uh, w_i2 + WD, e2n_in_b + Dd, nullptr, k2h, nullptr, 0, Dd, Dd);
  gemm16_kernel<<<dim3(Ee / 64, Dd / 64), gblk, 0, stream>>>(
      x1uh, w_i2 + 2 * WD, e2n_in_b + 2 * Dd, nullptr, nullptr, v2t, Ee, Dd, Dd);
  attn_kernel<<<dim3(Nn / 16, NHEADS), dim3(32), 0, stream>>>(
      q2h, k2h, v2t, Hm, atth, Ee, Ee, 1);
  gemm16_kernel<<<dim3(Nn / 64, Dd / 64), gblk, 0, stream>>>(
      atth, w_o2, e2n_out_b, tmp, nullptr, nullptr, 0, Dd, Dd);
  ln_residual_kernel<<<dim3(Nn / 8), cblk, 0, stream>>>(
      x0p, tmp, n2g, n2b, out, Nn == 0 ? x1uh : nullptr);
}
